// CAT_LSTM_20143396618775
// MI455X (gfx1250) — matrix operand order repaired and checked by reading, not compiled, not run on hardware
//
#include <hip/hip_runtime.h>
#include <hip/hip_bf16.h>
#include <stdint.h>

typedef __attribute__((ext_vector_type(16))) _Float16 v16h;
typedef __attribute__((ext_vector_type(8)))  float    v8f;

#define BATCH 128
#define SEQ   256
#define HID   256

// ---------------------------------------------------------------------------
// Weight convert + transpose: W[K x N] f32 (row-major) -> Wt[N x Kpad] f16,
// zero padded in K so GEMM B-tile loads are unguarded 16B vectors.
// ---------------------------------------------------------------------------
__global__ void wconv_kernel(const float* __restrict__ W, _Float16* __restrict__ Wt,
                             int K, int N, int Kpad) {
    int idx = blockIdx.x * blockDim.x + threadIdx.x;
    if (idx >= N * Kpad) return;
    int n = idx / Kpad, k = idx - n * Kpad;
    Wt[idx] = (k < K) ? (_Float16)W[(size_t)k * N + n] : (_Float16)0.0f;
}

// ---------------------------------------------------------------------------
// Activation convert + K-pad: A[M x K] f32 -> Ah[M x Kpad] f16 (zero padded).
// ---------------------------------------------------------------------------
__global__ void aconv_kernel(const float* __restrict__ A, _Float16* __restrict__ Ah,
                             int K, int Kpad, int total) {
    int idx = blockIdx.x * blockDim.x + threadIdx.x;
    if (idx >= total) return;
    int m = idx / Kpad, k = idx - m * Kpad;
    Ah[idx] = (k < K) ? (_Float16)A[(size_t)m * K + k] : (_Float16)0.0f;
}

// ---------------------------------------------------------------------------
// Strided activation convert: A[M x K] f32 -> Ah[m*ldo + k] f16 (no pad).
// ---------------------------------------------------------------------------
__global__ void aconv_stride_kernel(const float* __restrict__ A, _Float16* __restrict__ Ah,
                                    int K, int ldo, int total) {
    int idx = blockIdx.x * blockDim.x + threadIdx.x;
    if (idx >= total) return;
    int m = idx / K, k = idx - m * K;
    Ah[(size_t)m * ldo + k] = (_Float16)A[idx];
}

// ---------------------------------------------------------------------------
// WMMA GEMM:  C[M,N] = act( A @ B + bias [+ D] )
//   AMODE 0: A = A0 (f32, stride lda0): sync path, float4 loads + cvt.
//   AMODE 1: A = A1 (f16, stride lda1): double-buffered LDS filled with
//            global_load_async_to_lds_b128 (ASYNCcnt), copy overlaps WMMA.
//   B: pre-transposed f16 [N][Kpad], zero padded.  K % 32 == 0, N % 128 == 0.
//   D: optional f16 addend, row stride ldd.  Output stride ldc.
//   Block: 256 threads = 8 waves; tile 64x128, BK=32.
//   Wave (wm,wn): 16-row strip x 64-col strip -> FOUR 16x16 accums/wave.
// ---------------------------------------------------------------------------
template<int AMODE>
__global__ __launch_bounds__(256)
void wmma_gemm_kernel(const float* __restrict__ A0, int lda0,
                      const _Float16* __restrict__ A1, int lda1,
                      const _Float16* __restrict__ Bt, int kpad,
                      const float* __restrict__ bias,
                      const _Float16* __restrict__ Dadd, int ldd,
                      float* __restrict__ Cf, _Float16* __restrict__ Ch,
                      int ldc, int K, int relu)
{
    __shared__ _Float16 lds_a[2][64][40];    // (m,k); 80B row stride (16B aligned)
    __shared__ _Float16 lds_b[2][128][40];   // (n,k)

    const int tid  = threadIdx.x;
    const int lane = tid & 31;
    const int wave = tid >> 5;
    const int wm   = wave & 3;      // 16-row strip within 64
    const int wn   = wave >> 2;     // 64-col half within 128
    const int row0 = blockIdx.y * 64;
    const int col0 = blockIdx.x * 128;

    // staging coords: A 64x32 (1 b128/thread), B 128x32 (2 b128/thread)
    const int sa_r = tid >> 2, sa_k = (tid & 3) * 8;
    const int sb_r = tid >> 1, sb_k = (tid & 1) * 16;

    v8f acc[4] = {{}, {}, {}, {}};

    const int ktiles = K >> 5;

    // ---- per-wave compute: 1 A-frag, 4 B-frags, 4 WMMAs on buffer `b` ----
    auto compute_tile = [&](int b) {
        const int hi   = (lane < 16) ? 0 : 1;
        const int arow = wm * 16 + (lane & 15);
        union { uint32_t u[8]; v16h h; } fa;
        #pragma unroll
        for (int v = 0; v < 8; ++v) {
            int k = (hi ? 16 : 0) + 2 * v;
            fa.u[v] = *(const uint32_t*)&lds_a[b][arow][k];
        }
        const int bn = wn * 64 + (lane & 15);
        #pragma unroll
        for (int t = 0; t < 4; ++t) {
            union { uint32_t u[8]; v16h h; } fb;
            #pragma unroll
            for (int v = 0; v < 8; ++v) {
                int k = (hi ? 16 : 0) + 2 * v;
                fb.u[v] = *(const uint32_t*)&lds_b[b][bn + t * 16][k];
            }
            acc[t] = __builtin_amdgcn_wmma_f32_16x16x32_f16(false, fa.h, false, fb.h,
                                                            (short)0, acc[t], false, false);
        }
    };

    if constexpr (AMODE == 1) {
        // ---- async double-buffered pipeline (CDNA5 ASYNCcnt path) ----
        const _Float16* a_base = A1 + (size_t)(row0 + sa_r) * lda1 + sa_k;
        const _Float16* b_base = Bt + (size_t)(col0 + sb_r) * kpad + sb_k;
        auto issue_tile = [&](int kt, int b) {
            const _Float16* sa = a_base + (kt << 5);
            uint32_t la = (uint32_t)(uintptr_t)&lds_a[b][sa_r][sa_k];
            asm volatile("global_load_async_to_lds_b128 %0, %1, off"
                         :: "v"(la), "v"(sa) : "memory");
            const _Float16* sb = b_base + (kt << 5);
            uint32_t lb = (uint32_t)(uintptr_t)&lds_b[b][sb_r][sb_k];
            asm volatile("global_load_async_to_lds_b128 %0, %1, off"
                         :: "v"(lb), "v"(sb) : "memory");
            asm volatile("global_load_async_to_lds_b128 %0, %1, off"
                         :: "v"(lb + 16), "v"(sb + 8) : "memory");
        };
        issue_tile(0, 0);
        // steady state: prefetch kt+1 while computing kt (no in-loop branch)
        for (int kt = 0; kt < ktiles - 1; ++kt) {
            issue_tile(kt + 1, (kt + 1) & 1);
            asm volatile("s_wait_asynccnt 3" ::: "memory");   // tile kt resident
            __syncthreads();
            compute_tile(kt & 1);
            __syncthreads();
        }
        // peeled last tile
        asm volatile("s_wait_asynccnt 0" ::: "memory");
        __syncthreads();
        compute_tile((ktiles - 1) & 1);
    } else {
        // ---- sync f32 path (small latency-bound scan GEMMs) ----
        for (int kt = 0; kt < ktiles; ++kt) {
            const int k0 = kt << 5;
            {
                union { uint4 q; _Float16 h[8]; } pk;
                const float* src = A0 + (size_t)(row0 + sa_r) * lda0 + k0 + sa_k;
                float4 a = *(const float4*)src;
                float4 b = *(const float4*)(src + 4);
                pk.h[0] = (_Float16)a.x; pk.h[1] = (_Float16)a.y;
                pk.h[2] = (_Float16)a.z; pk.h[3] = (_Float16)a.w;
                pk.h[4] = (_Float16)b.x; pk.h[5] = (_Float16)b.y;
                pk.h[6] = (_Float16)b.z; pk.h[7] = (_Float16)b.w;
                *(uint4*)&lds_a[0][sa_r][sa_k] = pk.q;
            }
            {
                const _Float16* src = Bt + (size_t)(col0 + sb_r) * kpad + k0 + sb_k;
                *(uint4*)&lds_b[0][sb_r][sb_k]     = *(const uint4*)src;
                *(uint4*)&lds_b[0][sb_r][sb_k + 8] = *(const uint4*)(src + 8);
            }
            __syncthreads();
            compute_tile(0);
            __syncthreads();
        }
    }

    // ---- epilogue: C/D layout => VGPR r holds M=r (lanes 0-15) / M=8+r ----
    const int rbase = row0 + wm * 16 + ((lane < 16) ? 0 : 8);
    #pragma unroll
    for (int t = 0; t < 4; ++t) {
        const int cb = col0 + wn * 64 + t * 16 + (lane & 15);
        const float bb = bias[cb];
        #pragma unroll
        for (int r = 0; r < 8; ++r) {
            int gr = rbase + r;
            float v = acc[t][r] + bb;
            if (Dadd) v += (float)Dadd[(size_t)gr * ldd + cb];
            if (relu) v = fmaxf(v, 0.0f);
            if (Cf) Cf[(size_t)gr * ldc + cb] = v;
            else    Ch[(size_t)gr * ldc + cb] = (_Float16)v;
        }
    }
}

// ---------------------------------------------------------------------------
// Per-step TLSTM pointwise update.
// ---------------------------------------------------------------------------
__device__ __forceinline__ float sigm(float x) { return 1.0f / (1.0f + __expf(-x)); }

__global__ __launch_bounds__(256)
void lstm_pointwise_kernel(const float* __restrict__ G,
                           const float* __restrict__ CS1raw,
                           const float* __restrict__ ts, int s,
                           float* __restrict__ Cst, float* __restrict__ Hst,
                           float* __restrict__ feat)
{
    int idx = blockIdx.x * blockDim.x + threadIdx.x;   // 128*256
    int b = idx >> 8, j = idx & 255;
    float f  = sigm(G[b * 1024 + j]);
    float i  = sigm(G[b * 1024 + 256 + j]);
    float o  = sigm(G[b * 1024 + 512 + j]);
    float ct = sigm(G[b * 1024 + 768 + j]);
    float cs1   = tanhf(CS1raw[idx]);
    float tt    = ts[b * SEQ + s];
    float c_old = Cst[idx];
    float c_adj = (c_old - cs1) + cs1 * tt;
    float c_new = f * c_adj + i * ct;
    float h_new = o * tanhf(c_new);
    Cst[idx] = c_new;
    Hst[idx] = h_new;
    feat[(size_t)(b * SEQ + s) * HID + j] = h_new;
}

// ---------------------------------------------------------------------------
// Classifier: out[m] = feat[m,:] . w + b   (one wave32 per row)
// ---------------------------------------------------------------------------
__global__ __launch_bounds__(256)
void classifier_kernel(const float* __restrict__ feat, const float* __restrict__ w,
                       const float* __restrict__ bptr, float* __restrict__ out)
{
    int wave = threadIdx.x >> 5;
    int lane = threadIdx.x & 31;
    int row  = blockIdx.x * 8 + wave;
    const float* fr = feat + (size_t)row * HID;
    float s = 0.0f;
    #pragma unroll
    for (int k = lane; k < HID; k += 32) s += fr[k] * w[k];
    #pragma unroll
    for (int off = 16; off > 0; off >>= 1) s += __shfl_xor(s, off, 32);
    if (lane == 0) out[row] = s + bptr[0];
}

// ---------------------------------------------------------------------------
extern "C" void kernel_launch(void* const* d_in, const int* in_sizes, int n_in,
                              void* d_out, int out_size, void* d_ws, size_t ws_size,
                              hipStream_t stream) {
    (void)in_sizes; (void)n_in; (void)out_size; (void)ws_size;

    const float* x       = (const float*)d_in[0];    // [128,256,1024]
    const float* stages  = (const float*)d_in[1];    // [128,256,67]
    const float* ts      = (const float*)d_in[2];    // [128,256]
    const float* se_w1   = (const float*)d_in[3];    // [67,256]
    const float* se_b1   = (const float*)d_in[4];
    const float* se_w2   = (const float*)d_in[5];    // [256,512]
    const float* se_b2   = (const float*)d_in[6];
    const float* se_w3   = (const float*)d_in[7];    // [512,1024]
    const float* se_b3   = (const float*)d_in[8];
    const float* W_all_w = (const float*)d_in[9];    // [256,1024]
    const float* W_all_b = (const float*)d_in[10];
    const float* U_all_w = (const float*)d_in[11];   // [2048,1024]
    const float* U_all_b = (const float*)d_in[12];
    const float* W_d_w   = (const float*)d_in[13];   // [256,256]
    const float* W_d_b   = (const float*)d_in[14];
    const float* cls_w   = (const float*)d_in[15];   // [256,1]
    const float* cls_b   = (const float*)d_in[16];

    const int M  = BATCH * SEQ;                      // 32768
    float* out_cls  = (float*)d_out;                 // [32768]
    float* out_feat = (float*)d_out + M;             // [32768,256]

    // ---- workspace carve-up ----
    uint8_t* wsp = (uint8_t*)d_ws;
    auto carve = [&](size_t bytes) {
        uint8_t* p = wsp;
        wsp += (bytes + 255) & ~(size_t)255;
        return p;
    };
    _Float16* wt_se1  = (_Float16*)carve((size_t)256  * 96   * 2);
    _Float16* wt_se2  = (_Float16*)carve((size_t)512  * 256  * 2);
    _Float16* wt_se3  = (_Float16*)carve((size_t)1024 * 512  * 2);
    _Float16* wt_u    = (_Float16*)carve((size_t)1024 * 2048 * 2);
    _Float16* wt_wall = (_Float16*)carve((size_t)1024 * 256  * 2);
    _Float16* wt_wd   = (_Float16*)carve((size_t)256  * 256  * 2);
    _Float16* st_pad  = (_Float16*)carve((size_t)M * 96   * 2);   // padded stages f16
    _Float16* m1      = (_Float16*)carve((size_t)M * 256  * 2);
    _Float16* m2      = (_Float16*)carve((size_t)M * 512  * 2);
    _Float16* xin16   = (_Float16*)carve((size_t)M * 2048 * 2);   // [x | st] f16
    _Float16* u16     = (_Float16*)carve((size_t)M * 1024 * 2);   // U-projection out
    float*    Gbuf    = (float*)   carve((size_t)BATCH * 1024 * 4);
    float*    cs1raw  = (float*)   carve((size_t)BATCH * 256  * 4);
    float*    hbuf    = (float*)   carve((size_t)BATCH * 256  * 4);
    float*    cbuf    = (float*)   carve((size_t)BATCH * 256  * 4);

    // ---- convert/transpose weights to f16 [N][Kpad] ----
    auto wconv = [&](const float* W, _Float16* Wt, int K, int N, int Kpad) {
        int tot = N * Kpad;
        wconv_kernel<<<(tot + 255) / 256, 256, 0, stream>>>(W, Wt, K, N, Kpad);
    };
    wconv(se_w1,   wt_se1,  67,   256,  96);
    wconv(se_w2,   wt_se2,  256,  512,  256);
    wconv(se_w3,   wt_se3,  512,  1024, 512);
    wconv(U_all_w, wt_u,    2048, 1024, 2048);
    wconv(W_all_w, wt_wall, 256,  1024, 256);
    wconv(W_d_w,   wt_wd,   256,  256,  256);

    // ---- pad+convert stages: [M,67] f32 -> [M,96] f16 ----
    {
        int tot = M * 96;
        aconv_kernel<<<(tot + 255) / 256, 256, 0, stream>>>(stages, st_pad, 67, 96, tot);
    }
    // ---- convert x into columns 0..1023 of xin16 [M,2048] ----
    {
        int tot = M * 1024;
        aconv_stride_kernel<<<(tot + 255) / 256, 256, 0, stream>>>(x, xin16, 1024, 2048, tot);
    }

    // ---- stage-embedding MLP (32768 rows), async f16 GEMMs (AMODE=1) ----
    // L1: [M,96] @ [96,256] + b, ReLU -> m1 (f16)
    wmma_gemm_kernel<1><<<dim3(256 / 128, M / 64), 256, 0, stream>>>(
        nullptr, 0, st_pad, 96, wt_se1, 96, se_b1,
        nullptr, 0, nullptr, m1, 256, 96, 1);
    // L2: [M,256] @ [256,512] + b, ReLU -> m2 (f16)
    wmma_gemm_kernel<1><<<dim3(512 / 128, M / 64), 256, 0, stream>>>(
        nullptr, 0, m1, 256, wt_se2, 256, se_b2,
        nullptr, 0, nullptr, m2, 512, 256, 1);
    // L3: [M,512] @ [512,1024] + b, ReLU -> xin16 columns 1024..2047 (ldc=2048)
    wmma_gemm_kernel<1><<<dim3(1024 / 128, M / 64), 256, 0, stream>>>(
        nullptr, 0, m2, 512, wt_se3, 512, se_b3,
        nullptr, 0, nullptr, xin16 + 1024, 2048, 512, 1);

    // ---- U projection: xin16 @ U_all + b -> u16 (f16), K=2048 ----
    wmma_gemm_kernel<1><<<dim3(1024 / 128, M / 64), 256, 0, stream>>>(
        nullptr, 0, xin16, 2048, wt_u, 2048, U_all_b,
        nullptr, 0, nullptr, u16, 1024, 2048, 0);

    // ---- init h, c ----
    hipMemsetAsync(hbuf, 0, (size_t)BATCH * 256 * 4, stream);
    hipMemsetAsync(cbuf, 0, (size_t)BATCH * 256 * 4, stream);

    // ---- sequential TLSTM scan (sync f32 A, AMODE=0) ----
    for (int s = 0; s < SEQ; ++s) {
        // gates raw: h @ W_all + b + u[:,s,:]  -> G [128,1024] f32
        wmma_gemm_kernel<0><<<dim3(1024 / 128, BATCH / 64), 256, 0, stream>>>(
            hbuf, 256, nullptr, 0, wt_wall, 256, W_all_b,
            u16 + (size_t)s * 1024, SEQ * 1024, Gbuf, nullptr, 1024, 256, 0);
        // time-decay raw: c @ W_d + b -> cs1raw [128,256] f32
        wmma_gemm_kernel<0><<<dim3(256 / 128, BATCH / 64), 256, 0, stream>>>(
            cbuf, 256, nullptr, 0, wt_wd, 256, W_d_b,
            nullptr, 0, cs1raw, nullptr, 256, 256, 0);
        // pointwise cell/hidden update, write feat slice
        lstm_pointwise_kernel<<<(BATCH * 256) / 256, 256, 0, stream>>>(
            Gbuf, cs1raw, ts, s, cbuf, hbuf, out_feat);
    }

    // ---- classifier ----
    classifier_kernel<<<M / 8, 256, 0, stream>>>(out_feat, cls_w, cls_b, out_cls);
}
